// AttnBlock_55130200211965
// MI455X (gfx1250) — matrix operand order repaired and checked by reading, not compiled, not run on hardware
//
#include <hip/hip_runtime.h>
#include <hip/hip_bf16.h>

// ---------------- types ----------------
typedef __attribute__((ext_vector_type(16))) _Float16 v16h;
typedef __attribute__((ext_vector_type(8)))  _Float16 v8h;
typedef __attribute__((ext_vector_type(8)))  float    v8f;
typedef __attribute__((ext_vector_type(4)))  unsigned v4u;
typedef __attribute__((ext_vector_type(8)))  int      v8i;
typedef __attribute__((ext_vector_type(4)))  int      v4i;

static __device__ __forceinline__ v16h cat8(v8h lo, v8h hi) {
    return __builtin_shufflevector(lo, hi, 0,1,2,3,4,5,6,7,8,9,10,11,12,13,14,15);
}

// async copy 16B global -> LDS (per lane), tracked by ASYNCcnt
static __device__ __forceinline__ void async_cp16(const void* gptr, void* lptr) {
    unsigned lds = (unsigned)(uintptr_t)lptr;     // low 32 bits of generic = LDS byte addr
    asm volatile("global_load_async_to_lds_b128 %0, %1, off"
                 :: "v"(lds), "v"(gptr) : "memory");
}
static __device__ __forceinline__ void async_wait(int n) {
    if (n == 0)      asm volatile("s_wait_asynccnt 0x0" ::: "memory");
    else             asm volatile("s_wait_asynccnt 0x8" ::: "memory");
}

// TDM: load a 2-D tile (tile_d1 rows x tile_d0 elems, 2-byte elems) of a row-major
// tensor [tensor_d1, tensor_d0] into LDS, optional per-row LDS padding.
static __device__ __forceinline__ void tdm_load_2d(const void* gptr, void* lptr,
        unsigned tensor_d0, unsigned tensor_d1, unsigned tile_d0, unsigned tile_d1,
        unsigned stride0, unsigned pad_interval, unsigned pad_amount, bool pad_en) {
#if __has_builtin(__builtin_amdgcn_tensor_load_to_lds)
    unsigned long long ga = (unsigned long long)(uintptr_t)gptr;
    unsigned lds = (unsigned)(uintptr_t)lptr;
    v4u g0 = { 1u,                                             // count=1 (valid descriptor)
               lds,                                            // lds_addr
               (unsigned)(ga & 0xFFFFFFFFu),                   // global_addr[31:0]
               (unsigned)(((ga >> 32) & 0x01FFFFFFu) | 0x80000000u) }; // addr[56:32] | type=2
    unsigned w0 = (1u << 16)                                   // data_size = 2 bytes
                | (pad_en ? (1u << 20) : 0u)                   // pad_enable
                | (pad_interval << 22) | (pad_amount << 25);
    v8i g1 = { (int)w0,
               (int)((tensor_d0 & 0xFFFFu) << 16),                                  // dim0[15:0]
               (int)(((tensor_d0 >> 16) & 0xFFFFu) | ((tensor_d1 & 0xFFFFu) << 16)),// dim0 hi | dim1 lo
               (int)(((tensor_d1 >> 16) & 0xFFFFu) | ((tile_d0 & 0xFFFFu) << 16)),  // dim1 hi | tile0
               (int)(tile_d1 & 0xFFFFu),                                            // tile1 | tile2=0
               (int)stride0, 0, 0 };                                                // dim0_stride
    v4i z4 = {0,0,0,0};
#if defined(__clang_major__) && __clang_major__ >= 23
    v8i z8 = {0,0,0,0,0,0,0,0};
    __builtin_amdgcn_tensor_load_to_lds(g0, g1, z4, z4, z8, 0);
#else
    __builtin_amdgcn_tensor_load_to_lds(g0, g1, z4, z4, 0);
#endif
#if __has_builtin(__builtin_amdgcn_s_wait_tensorcnt)
    __builtin_amdgcn_s_wait_tensorcnt((short)0);
#endif
#else
    // fallback: wave-cooperative async copy (lane strided over columns of each row)
    (void)stride0; (void)pad_interval; (void)pad_amount; (void)pad_en;
    const int lane = threadIdx.x & 31;
    const unsigned ldrow = tile_d0 + (pad_en ? 8u : 0u);
    for (unsigned r = 0; r < tile_d1; ++r)
        for (unsigned c = lane * 8; c < tile_d0; c += 32 * 8)
            async_cp16((const _Float16*)gptr + (size_t)r * tensor_d0 + c,
                       (_Float16*)lptr + (size_t)r * ldrow + c);
    asm volatile("s_wait_asynccnt 0x0" ::: "memory");
#endif
}

// ---------------- problem constants ----------------
#define CC   512          // channels
#define GG   32           // groups
#define CPG  16           // channels per group
#define NN   4096         // 64*64 spatial
#define BB   2            // batch

// GEMM tiling
#define BM 128
#define BN 128
#define BK 64
#define LDT 72            // LDS row stride (halves) for 64-wide K tiles

// ---------------- kernel 0: fp32 -> f16 weight conversion ----------------
__global__ __launch_bounds__(256)
void cvt_weights(const float* __restrict__ wq, const float* __restrict__ wk,
                 const float* __restrict__ wv, const float* __restrict__ wp,
                 _Float16* __restrict__ out) {
    int i = blockIdx.x * 256 + threadIdx.x;            // 4 * C*C total
    int which = i >> 18;                               // C*C = 262144 = 2^18
    int off   = i & ((1 << 18) - 1);
    const float* src = (which == 0) ? wq : (which == 1) ? wk : (which == 2) ? wv : wp;
    out[i] = (_Float16)src[off];
}

// ---------------- kernel 1: GroupNorm -> f16 activations, TRANSPOSED [B,N,C] ----------------
__global__ __launch_bounds__(256)
void groupnorm_kernel(const float* __restrict__ x, const float* __restrict__ gsc,
                      const float* __restrict__ gbs, _Float16* __restrict__ hT) {
    const int b = blockIdx.x >> 5;
    const int g = blockIdx.x & 31;
    const int tid = threadIdx.x;
    const size_t base = ((size_t)(b * CC + g * CPG)) * NN;
    const float* xp = x + base;

    __shared__ float s1[256];
    __shared__ float s2[256];
    float a = 0.f, a2 = 0.f;
    for (int i = tid; i < CPG * NN; i += 256) {
        float v = xp[i];
        a += v; a2 += v * v;
    }
    s1[tid] = a; s2[tid] = a2;
    __syncthreads();
    for (int off = 128; off > 0; off >>= 1) {
        if (tid < off) { s1[tid] += s1[tid + off]; s2[tid] += s2[tid + off]; }
        __syncthreads();
    }
    const float invn = 1.f / (float)(CPG * NN);
    const float mean = s1[0] * invn;
    const float var  = s2[0] * invn - mean * mean;
    const float inv  = rsqrtf(var + 1e-6f);

    _Float16* hp = hT + (size_t)b * NN * CC;           // [N, C]
    for (int i = tid; i < CPG * NN; i += 256) {
        int c = g * CPG + (i >> 12);
        int n = i & (NN - 1);
        float v = (xp[i] - mean) * inv * gsc[c] + gbs[c];
        hp[(size_t)n * CC + c] = (_Float16)v;
    }
}

// ---------------- kernel 2: WMMA "TN" GEMM, double-buffered async staging ----------------
// out[m,n] = sum_k A[m,k] * Bt[n,k]   (A: [M,K] f16, Bt: [Ncols,K] f16, both K-contiguous)
// MODE 0: f16 [M,N] store; MODE 1: f16 transposed [N,M] store; MODE 2: f32 + bias + resid.
static __device__ __forceinline__ void gemm_stage(const _Float16* A, const _Float16* Bt,
                                                  _Float16* As, _Float16* Bs,
                                                  int tid, int gm0, int gn0, int K, int k0) {
#pragma unroll
    for (int rep = 0; rep < 4; ++rep) {
        int idx = tid + rep * 256;
        int r = idx >> 3, c8 = (idx & 7) * 8;
        async_cp16(A  + (size_t)(gm0 + r) * K + k0 + c8, &As[r * LDT + c8]);
        async_cp16(Bt + (size_t)(gn0 + r) * K + k0 + c8, &Bs[r * LDT + c8]);
    }
}

template<int MODE>
__global__ __launch_bounds__(256)
void gemm_wmma(const _Float16* __restrict__ Ag, const _Float16* __restrict__ Btg,
               const float* __restrict__ bias, const float* __restrict__ resid,
               void* __restrict__ outp,
               int M, int Nc, int K, float scale,
               long long strideA, long long strideB, long long strideO) {
    extern __shared__ char gsmem[];
    _Float16* AsBuf = (_Float16*)gsmem;            // [2][BM][LDT]
    _Float16* BsBuf = AsBuf + 2 * BM * LDT;        // [2][BN][LDT]

    const int tid  = threadIdx.x;
    const int wid  = tid >> 5;
    const int lane = tid & 31;
    const int lr   = lane & 15;
    const int hi   = lane >> 4;
    const int wm   = wid & 3;          // 4 waves along M (4*32 = 128)
    const int wn   = wid >> 2;         // 2 waves along N (2*64 = 128)
    const int z    = blockIdx.z;

    const _Float16* A  = Ag  + (size_t)z * strideA;
    const _Float16* Bt = Btg + (size_t)z * strideB;
    const int gm0 = blockIdx.y * BM;
    const int gn0 = blockIdx.x * BN;

    v8f acc[2][4];
#pragma unroll
    for (int i = 0; i < 2; ++i)
#pragma unroll
        for (int j = 0; j < 4; ++j)
            acc[i][j] = (v8f){0.f,0.f,0.f,0.f,0.f,0.f,0.f,0.f};

    const int T = K / BK;
    gemm_stage(A, Bt, AsBuf, BsBuf, tid, gm0, gn0, K, 0);

    for (int t = 0; t < T; ++t) {
        const int buf = t & 1;
        if (t + 1 < T)
            gemm_stage(A, Bt, AsBuf + ((t + 1) & 1) * BM * LDT,
                       BsBuf + ((t + 1) & 1) * BN * LDT, tid, gm0, gn0, K, (t + 1) * BK);
        async_wait(t + 1 < T ? 8 : 0);      // tile t resident (loads retire in order)
        __syncthreads();

        const _Float16* As = AsBuf + buf * BM * LDT;
        const _Float16* Bs = BsBuf + buf * BN * LDT;
#pragma unroll
        for (int kk = 0; kk < BK; kk += 32) {
            v16h afr[2], bfr[4];
#pragma unroll
            for (int tm = 0; tm < 2; ++tm) {
                const _Float16* ap = &As[(wm * 32 + tm * 16 + lr) * LDT + kk + hi * 8];
                afr[tm] = cat8(*(const v8h*)ap, *(const v8h*)(ap + 16));
            }
#pragma unroll
            for (int tn = 0; tn < 4; ++tn) {
                const _Float16* bp = &Bs[(wn * 64 + tn * 16 + lr) * LDT + kk + hi * 8];
                bfr[tn] = cat8(*(const v8h*)bp, *(const v8h*)(bp + 16));
            }
#pragma unroll
            for (int tm = 0; tm < 2; ++tm)
#pragma unroll
                for (int tn = 0; tn < 4; ++tn)
                    acc[tm][tn] = __builtin_amdgcn_wmma_f32_16x16x32_f16(
                        false, afr[tm], false, bfr[tn], (short)0, acc[tm][tn], false, false);
        }
        __syncthreads();   // everyone done reading buf before it is overwritten
    }

    // epilogue
#pragma unroll
    for (int tm = 0; tm < 2; ++tm) {
        const int gm = gm0 + wm * 32 + tm * 16 + hi * 8;
#pragma unroll
        for (int tn = 0; tn < 4; ++tn) {
            const int gn = gn0 + wn * 64 + tn * 16 + lr;
            if (MODE == 0) {
                _Float16* out = (_Float16*)outp + (size_t)z * strideO;
#pragma unroll
                for (int r = 0; r < 8; ++r) {
                    float bv = bias ? bias[gm + r] : 0.f;
                    out[(size_t)(gm + r) * Nc + gn] = (_Float16)((acc[tm][tn][r] + bv) * scale);
                }
            } else if (MODE == 1) {
                _Float16* out = (_Float16*)outp + (size_t)z * strideO;
                v8h pk;
#pragma unroll
                for (int r = 0; r < 8; ++r) {
                    float bv = bias ? bias[gm + r] : 0.f;
                    pk[r] = (_Float16)((acc[tm][tn][r] + bv) * scale);
                }
                *(v8h*)&out[(size_t)gn * M + gm] = pk;
            } else {
                float* out = (float*)outp + (size_t)z * strideO;
                const float* res = resid + (size_t)z * strideO;
#pragma unroll
                for (int r = 0; r < 8; ++r) {
                    float bv = bias ? bias[gm + r] : 0.f;
                    size_t idx = (size_t)(gm + r) * Nc + gn;
                    out[idx] = acc[tm][tn][r] + bv + res[idx];
                }
            }
        }
    }
}

// ---------------- kernel 3: attention scores + exact softmax ----------------
// Per block: 16 query rows. q tile (16x512) loaded once via TDM (padded rows),
// k tiles (256 j-rows x 64 k) double-buffered via async copies.
// Full 16x4096 score row in LDS (f16), exact 2-pass softmax -> P[b,i,j] f16.
#define JT 256
__global__ __launch_bounds__(256)
void attn_scores_softmax(const _Float16* __restrict__ qT,   // [B,N,C]
                         const _Float16* __restrict__ kT,   // [B,N,C]
                         _Float16* __restrict__ P) {        // [B,N,N]
    extern __shared__ char smem[];
    _Float16* qTs = (_Float16*)smem;               // [16][520]
    _Float16* ks  = qTs + 16 * 520;                // [2][JT][LDT]
    _Float16* S   = ks + 2 * JT * LDT;             // [16][4096]
    float* red    = (float*)(S + 16 * NN);         // [16][17]
    float* rowst  = red + 16 * 17;                 // [32]: max, then inv-sum

    const int tid  = threadIdx.x;
    const int wid  = tid >> 5;
    const int lane = tid & 31;
    const int lr   = lane & 15;
    const int hi   = lane >> 4;
    const int b    = blockIdx.y;
    const int i0   = blockIdx.x * 16;

    const _Float16* q = qT + ((size_t)b * NN + i0) * CC;
    const _Float16* k = kT + (size_t)b * NN * CC;

    // q tile via Tensor Data Mover: 16 rows x 512 halves, LDS rows padded to 520
    // (pad_interval=7 -> every 256 DWORDs = 512 halves; pad_amount=3 -> 4 DWORDs = 8 halves)
    if (wid == 0)
        tdm_load_2d(q, qTs, CC, 16, CC, 16, CC, 7u, 3u, true);
    __syncthreads();

    // pipelined (jt, kt) loop: 16 j-tiles x 8 k-steps
    const int NSTAGE = (NN / JT) * (CC / BK);      // 128
    {
        // prologue stage 0
#pragma unroll
        for (int rep = 0; rep < 8; ++rep) {
            int idx = tid + rep * 256;
            int r = idx >> 3, c8 = (idx & 7) * 8;
            async_cp16(k + (size_t)r * CC + c8, &ks[r * LDT + c8]);
        }
    }
    v8f acc[2];
    for (int s = 0; s < NSTAGE; ++s) {
        const int kt = s & 7;                      // k-step within j-tile
        const int jt = s >> 3;
        if (kt == 0) {
            acc[0] = (v8f){0.f,0.f,0.f,0.f,0.f,0.f,0.f,0.f};
            acc[1] = (v8f){0.f,0.f,0.f,0.f,0.f,0.f,0.f,0.f};
        }
        if (s + 1 < NSTAGE) {
            const int nkt = (s + 1) & 7, njt = (s + 1) >> 3;
            _Float16* dst = ks + ((s + 1) & 1) * JT * LDT;
#pragma unroll
            for (int rep = 0; rep < 8; ++rep) {
                int idx = tid + rep * 256;
                int r = idx >> 3, c8 = (idx & 7) * 8;
                async_cp16(k + (size_t)(njt * JT + r) * CC + nkt * BK + c8,
                           &dst[r * LDT + c8]);
            }
        }
        async_wait(s + 1 < NSTAGE ? 8 : 0);
        __syncthreads();

        const _Float16* kb = ks + (s & 1) * JT * LDT;
#pragma unroll
        for (int kk = 0; kk < BK; kk += 32) {
            const _Float16* ap = &qTs[lr * 520 + kt * BK + kk + hi * 8];
            v16h a = cat8(*(const v8h*)ap, *(const v8h*)(ap + 16));
#pragma unroll
            for (int jn = 0; jn < 2; ++jn) {
                const _Float16* bp = &kb[(jn * 128 + wid * 16 + lr) * LDT + kk + hi * 8];
                v16h bf = cat8(*(const v8h*)bp, *(const v8h*)(bp + 16));
                acc[jn] = __builtin_amdgcn_wmma_f32_16x16x32_f16(
                    false, a, false, bf, (short)0, acc[jn], false, false);
            }
        }
        __syncthreads();

        if (kt == 7) {
#pragma unroll
            for (int jn = 0; jn < 2; ++jn) {
                const int j = jt * JT + jn * 128 + wid * 16 + lr;
#pragma unroll
                for (int r = 0; r < 8; ++r)
                    S[(hi * 8 + r) * NN + j] = (_Float16)acc[jn][r];
            }
        }
    }
    __syncthreads();

    // exact softmax over each of the 16 rows (16 threads per row)
    const int row = tid >> 4, t = tid & 15;
    float m = -1e30f;
    for (int idx = t; idx < NN; idx += 16)
        m = fmaxf(m, (float)S[row * NN + idx]);
    red[row * 17 + t] = m;
    __syncthreads();
    if (t == 0) {
        float mm = red[row * 17];
        for (int u = 1; u < 16; ++u) mm = fmaxf(mm, red[row * 17 + u]);
        rowst[row] = mm;
    }
    __syncthreads();
    const float rmax = rowst[row];
    float sum = 0.f;
    for (int idx = t; idx < NN; idx += 16)
        sum += __expf((float)S[row * NN + idx] - rmax);
    red[row * 17 + t] = sum;
    __syncthreads();
    if (t == 0) {
        float ss = 0.f;
        for (int u = 0; u < 16; ++u) ss += red[row * 17 + u];
        rowst[16 + row] = 1.f / ss;
    }
    __syncthreads();
    const float inv = rowst[16 + row];
    _Float16* Pp = P + ((size_t)b * NN + (i0 + row)) * (size_t)NN;
    for (int idx = t; idx < NN; idx += 16)
        Pp[idx] = (_Float16)(__expf((float)S[row * NN + idx] - rmax) * inv);
}

// ---------------- host-side orchestration ----------------
extern "C" void kernel_launch(void* const* d_in, const int* in_sizes, int n_in,
                              void* d_out, int out_size, void* d_ws, size_t ws_size,
                              hipStream_t stream) {
    const float* x   = (const float*)d_in[0];
    const float* gsc = (const float*)d_in[1];
    const float* gbs = (const float*)d_in[2];
    const float* wq  = (const float*)d_in[3];
    const float* bq  = (const float*)d_in[4];
    const float* wk  = (const float*)d_in[5];
    const float* bk  = (const float*)d_in[6];
    const float* wv  = (const float*)d_in[7];
    const float* bv  = (const float*)d_in[8];
    const float* wp  = (const float*)d_in[9];
    const float* bp  = (const float*)d_in[10];
    float* out = (float*)d_out;

    // workspace layout
    char* ws = (char*)d_ws;
    _Float16* wf16 = (_Float16*)ws;                 ws += (size_t)4 * CC * CC * 2;
    _Float16* hT   = (_Float16*)ws;                 ws += (size_t)BB * NN * CC * 2;
    _Float16* qT   = (_Float16*)ws;                 ws += (size_t)BB * NN * CC * 2;
    _Float16* kT   = (_Float16*)ws;                 ws += (size_t)BB * NN * CC * 2;
    _Float16* v    = (_Float16*)ws;                 ws += (size_t)BB * CC * NN * 2;
    _Float16* P    = (_Float16*)ws;                 ws += (size_t)BB * NN * (size_t)NN * 2;
    _Float16* OT   = (_Float16*)ws;                 ws += (size_t)BB * NN * CC * 2;

    cvt_weights<<<(4 * CC * CC) / 256, 256, 0, stream>>>(wq, wk, wv, wp, wf16);
    groupnorm_kernel<<<BB * GG, 256, 0, stream>>>(x, gsc, gbs, hT);

    const float qscale = 0.044194173824159216f;     // 512^-0.5
    const size_t gemm_smem = (size_t)2 * (BM + BN) * LDT * 2;   // 73,728 B
    dim3 gQKV(NN / BN, CC / BM, BB);                // (32, 4, 2)

    // q = (wq*h + bq) * scale -> qT[N,C]
    gemm_wmma<1><<<gQKV, 256, gemm_smem, stream>>>(wf16 + 0 * CC * CC, hT, bq, nullptr, qT,
                                                   CC, NN, CC, qscale,
                                                   0LL, (long long)NN * CC, (long long)NN * CC);
    // k = wk*h + bk -> kT[N,C]
    gemm_wmma<1><<<gQKV, 256, gemm_smem, stream>>>(wf16 + 1 * CC * CC, hT, bk, nullptr, kT,
                                                   CC, NN, CC, 1.f,
                                                   0LL, (long long)NN * CC, (long long)NN * CC);
    // v = wv*h + bv -> v[C,N]
    gemm_wmma<0><<<gQKV, 256, gemm_smem, stream>>>(wf16 + 2 * CC * CC, hT, bv, nullptr, v,
                                                   CC, NN, CC, 1.f,
                                                   0LL, (long long)NN * CC, (long long)CC * NN);

    // P = softmax(q . k^T)
    const size_t attn_smem = (size_t)(16 * 520 + 2 * JT * LDT + 16 * NN) * 2
                           + (size_t)(16 * 17 + 32) * 4;        // ~218 KB (<320 KB WGP LDS)
    attn_scores_softmax<<<dim3(NN / 16, BB), 256, attn_smem, stream>>>(qT, kT, P);

    // OT[i,c] = P[i,:] . v[c,:]
    gemm_wmma<0><<<dim3(CC / BN, NN / BM, BB), 256, gemm_smem, stream>>>(
        P, v, nullptr, nullptr, OT,
        NN, CC, NN, 1.f,
        (long long)NN * NN, (long long)CC * NN, (long long)NN * CC);

    // out = wp*O + bp + x (fp32)
    gemm_wmma<2><<<gQKV, 256, gemm_smem, stream>>>(wf16 + 3 * CC * CC, OT, bp, x, out,
                                                   CC, NN, CC, 1.f,
                                                   0LL, (long long)NN * CC, (long long)CC * NN);
}
